// FunnelGNN_EdgeAttr_44375602103092
// MI455X (gfx1250) — compile-verified, matrix operand order checked by reading, not run on hardware
//
#include <hip/hip_runtime.h>
#include <hip/hip_bf16.h>
#include <math.h>

// ---------------------------------------------------------------------------
// FunnelGNN_EdgeAttr for MI455X (gfx1250, wave32, WMMA bf16 + LDS-blocked GEMM)
// ---------------------------------------------------------------------------

typedef float  v8f   __attribute__((ext_vector_type(8)));
typedef __bf16 v8bf  __attribute__((ext_vector_type(8)));
typedef __bf16 v16bf __attribute__((ext_vector_type(16)));

constexpr int   Nn   = 50000;      // nodes
constexpr int   Ne   = 800000;     // edges
constexpr int   Ng   = 256;        // graphs
constexpr int   Hdim = 128;        // hidden
constexpr int   ZW   = 18 * Hdim;  // 2304 pooled feature width
constexpr float EPSF = 1e-5f;
constexpr float SLOPE = 0.01f;     // jax.nn.leaky_relu default

constexpr int NT_PER_BLK = 2;      // 16-col N tiles per block (32 cols)
constexpr int WAVES      = 8;      // wave32s (M tiles) per block
constexpr int KCH_MAX    = 8;      // max K/32 (K up to 256)

// ------------------------------ small utils -------------------------------

__global__ void zero_f32_kernel(float* p, long long n) {
  long long i = (long long)blockIdx.x * blockDim.x + threadIdx.x;
  if (i < n) p[i] = 0.0f;
}

__global__ void f32_to_bf16_kernel(const float* in, __bf16* out, long long n) {
  long long i = (long long)blockIdx.x * blockDim.x + threadIdx.x;
  if (i < n) out[i] = (__bf16)in[i];
}

// Pack W[K,Fout] (f32 row-major, fan_in x fan_out) into per-lane WMMA-B v16bf
// layout: Bp[((ntile*kchunks + kc)*32 + lane)*16 + s]
//   col = ntile*16 + (lane&15)
//   k   = kc*32 + (lane>>4)*8 + (s<8 ? s : (s-8)+16)
__global__ void pack_w_kernel(const float* W, __bf16* Bp, int K, int Fout) {
  int idx = blockIdx.x * blockDim.x + threadIdx.x;
  int kchunks = K >> 5;
  int total = (Fout >> 4) * kchunks * 32 * 16;
  if (idx >= total) return;
  int s     = idx & 15;
  int lane  = (idx >> 4) & 31;
  int kc    = (idx >> 9) % kchunks;
  int ntile = idx / (kchunks << 9);
  int col = ntile * 16 + (lane & 15);
  int k   = kc * 32 + ((lane >> 4) << 3) + (s < 8 ? s : (s - 8) + 16);
  Bp[idx] = (__bf16)W[(long long)k * Fout + col];
}

// ------------------------------ edge aggregation --------------------------

__global__ void edge_agg_kernel(const float* __restrict__ h,
                                const int* __restrict__ ei,
                                const float* __restrict__ ew,
                                float* __restrict__ agg, int F) {
  int tid = blockIdx.x * blockDim.x + threadIdx.x;
  int cpe = F >> 2;                       // float4 chunks per edge
  long long total = (long long)Ne * cpe;
  if (tid >= total) return;
  int e = tid / cpe;
  int c = (tid % cpe) << 2;
  int src = ei[e];
  int dst = ei[Ne + e];
  float w = ew[e];
  const float4 v = *(const float4*)(h + (long long)src * F + c);
  float* d = agg + (long long)dst * F + c;
  atomicAdd(d + 0, w * v.x);
  atomicAdd(d + 1, w * v.y);
  atomicAdd(d + 2, w * v.z);
  atomicAdd(d + 3, w * v.w);
}

// ------------------------------ fused WMMA GEMM ---------------------------
// Per block: stage packed rel/root weight fragments for a 32-column range in
// LDS once; 8 waves each compute a 16x32 output tile (2 WMMA pairs per K
// step), reading A from global (L2-resident) and B from LDS (ds_load_b128).
// out = leaky_relu( agg @ Wrel + h @ Wroot + bias )

__global__ __launch_bounds__(256)
void gconv_gemm_wmma(const __bf16* __restrict__ Abf,   // agg, row-major [Nn,K]
                     const __bf16* __restrict__ Hbf,   // root input, [Nn,K]
                     const __bf16* __restrict__ BpRel, // packed rel weights
                     const __bf16* __restrict__ BpRoot,// packed root weights
                     const float*  __restrict__ bias,
                     float* __restrict__ out,          // [Nn,Fout]
                     int K, int Fout, int mtiles) {
  const int kchunks = K >> 5;
  const int nt0 = blockIdx.y * NT_PER_BLK;
  const int tid = threadIdx.x;

  // worst case: 2 operands * 2 ntiles * 8 kchunks * 512 bf16 = 32 KB
  __shared__ __align__(32) __bf16 smem[2 * NT_PER_BLK * KCH_MAX * 512];
  __bf16* smRel  = smem;
  __bf16* smRoot = smem + NT_PER_BLK * kchunks * 512;

  {  // cooperative copy of this column-range's packed B fragments into LDS
    const long long base = (long long)nt0 * kchunks * 512;
    const int cnt16 = NT_PER_BLK * kchunks * 512 / 8;  // uint4 (8 bf16) chunks
    const uint4* gr = (const uint4*)(BpRel + base);
    const uint4* go = (const uint4*)(BpRoot + base);
    uint4* sr = (uint4*)smRel;
    uint4* so = (uint4*)smRoot;
    for (int i = tid; i < cnt16; i += 256) { sr[i] = gr[i]; so[i] = go[i]; }
  }
  __syncthreads();

  const int wave = tid >> 5;
  const int lane = tid & 31;
  const int mt = blockIdx.x * WAVES + wave;
  if (mt >= mtiles) return;   // whole-wave guard (EXEC stays all-ones)

  const int m  = lane & 15;
  const int kb = (lane >> 4) << 3;  // 0 or 8
  const __bf16* arow = Abf + (long long)(mt * 16 + m) * K;
  const __bf16* hrow = Hbf + (long long)(mt * 16 + m) * K;

  v8f acc[NT_PER_BLK] = {};

  for (int kc = 0; kc < kchunks; ++kc) {
    const int ko = kc * 32 + kb;
    // A operands: ISA 16-bit A layout (two contiguous 8-element halves)
    v8bf a_lo = *(const v8bf*)(arow + ko);
    v8bf a_hi = *(const v8bf*)(arow + ko + 16);
    v16bf a = __builtin_shufflevector(a_lo, a_hi, 0, 1, 2, 3, 4, 5, 6, 7,
                                      8, 9, 10, 11, 12, 13, 14, 15);
    v8bf h_lo = *(const v8bf*)(hrow + ko);
    v8bf h_hi = *(const v8bf*)(hrow + ko + 16);
    v16bf h = __builtin_shufflevector(h_lo, h_hi, 0, 1, 2, 3, 4, 5, 6, 7,
                                      8, 9, 10, 11, 12, 13, 14, 15);
#pragma unroll
    for (int t = 0; t < NT_PER_BLK; ++t) {
      v16bf brel  = *(const v16bf*)(smRel  + (((t * kchunks + kc) * 32 + lane) << 4));
      v16bf broot = *(const v16bf*)(smRoot + (((t * kchunks + kc) * 32 + lane) << 4));
      acc[t] = __builtin_amdgcn_wmma_f32_16x16x32_bf16(false, a, false, brel,
                                                       (short)0, acc[t], false, false);
      acc[t] = __builtin_amdgcn_wmma_f32_16x16x32_bf16(false, h, false, broot,
                                                       (short)0, acc[t], false, false);
    }
  }

  // C layout: VGPR i -> row = i + (lane>>4)*8 ; col = lane&15
  const int rbase = mt * 16 + ((lane >> 4) << 3);
#pragma unroll
  for (int t = 0; t < NT_PER_BLK; ++t) {
    const int col = (nt0 + t) * 16 + (lane & 15);
    const float bv = bias[col];
#pragma unroll
    for (int i = 0; i < 8; ++i) {
      float v = acc[t][i] + bv;
      v = v > 0.0f ? v : SLOPE * v;   // fused leaky_relu
      out[(long long)(rbase + i) * Fout + col] = v;
    }
  }
}

// ------------------------------ BatchNorm ---------------------------------

__global__ void bn_stats_kernel(const float* __restrict__ h, float* stats, int F) {
  __shared__ float ss[256];
  __shared__ float sq[256];
  int ch = blockIdx.x;
  int t  = threadIdx.x;
  float s = 0.0f, q = 0.0f;
  for (int r = t; r < Nn; r += 256) {
    float v = h[(long long)r * F + ch];
    s += v; q += v * v;
  }
  ss[t] = s; sq[t] = q;
  __syncthreads();
  for (int o = 128; o > 0; o >>= 1) {
    if (t < o) { ss[t] += ss[t + o]; sq[t] += sq[t + o]; }
    __syncthreads();
  }
  if (t == 0) { stats[ch] = ss[0]; stats[F + ch] = sq[0]; }
}

__global__ void bn_apply_kernel(float* __restrict__ h, const float* __restrict__ stats,
                                const float* __restrict__ gamma,
                                const float* __restrict__ beta, int F) {
  long long i = (long long)blockIdx.x * blockDim.x + threadIdx.x;
  long long total = (long long)Nn * F;
  if (i >= total) return;
  int ch = (int)(i % F);
  float mu  = stats[ch] / (float)Nn;
  float var = stats[F + ch] / (float)Nn - mu * mu;
  float v = h[i];
  h[i] = (v - mu) * rsqrtf(var + EPSF) * gamma[ch] + beta[ch];
}

// ------------------------------ pooling -----------------------------------

__device__ __forceinline__ int fkey(float f) {
  int i = __float_as_int(f);
  return i < 0 ? (i ^ 0x7fffffff) : i;
}
__device__ __forceinline__ float unkey(int i) {
  return __int_as_float(i < 0 ? (i ^ 0x7fffffff) : i);
}

__global__ void pool_init_kernel(float* psum, int* pmax, float* pcnt, int F) {
  int i = blockIdx.x * blockDim.x + threadIdx.x;
  int total = Ng * F;
  if (i < total) { psum[i] = 0.0f; pmax[i] = INT_MIN; }
  if (i < Ng) pcnt[i] = 0.0f;
}

__global__ void pool_acc_kernel(const float* __restrict__ h,
                                const int* __restrict__ batch,
                                float* psum, int* pmax, float* pcnt, int F) {
  long long i = (long long)blockIdx.x * blockDim.x + threadIdx.x;
  long long total = (long long)Nn * F;
  if (i >= total) return;
  int node = (int)(i / F);
  int f    = (int)(i % F);
  int g    = batch[node];
  float v  = h[i];
  atomicAdd(&psum[(long long)g * F + f], v);
  atomicMax(&pmax[(long long)g * F + f], fkey(v));
  if (f == 0) atomicAdd(&pcnt[g], 1.0f);
}

__global__ void pool_write_kernel(const float* psum, const int* pmax,
                                  const float* pcnt, float* z, int zoff, int F) {
  int i = blockIdx.x * blockDim.x + threadIdx.x;
  int total = Ng * F;
  if (i >= total) return;
  int g = i / F, f = i % F;
  float cnt = pcnt[g];
  float s   = psum[i];
  float mx  = (cnt > 0.0f) ? unkey(pmax[i]) : 0.0f;
  float mean = s / fmaxf(cnt, 1.0f);
  float* zr = z + (long long)g * ZW + zoff;
  zr[f]         = mx;
  zr[F + f]     = mean;
  zr[2 * F + f] = s;
}

// ------------------------------ FC head -----------------------------------

__global__ void fc_kernel(const float* __restrict__ in, const float* __restrict__ W,
                          const float* __restrict__ b, float* __restrict__ out,
                          int K, int Nout, int do_relu) {
  int idx = blockIdx.x * blockDim.x + threadIdx.x;
  if (idx >= Ng * Nout) return;
  int g = idx / Nout, j = idx % Nout;
  float acc = b[j];
  const float* row = in + (long long)g * K;
  for (int k = 0; k < K; ++k) acc += row[k] * W[(long long)k * Nout + j];
  if (do_relu) acc = fmaxf(acc, 0.0f);
  out[idx] = acc;
}

__global__ void head_final_kernel(const float* __restrict__ in,
                                  const float* __restrict__ W,
                                  const float* __restrict__ b,
                                  float* __restrict__ out) {
  int g = blockIdx.x * blockDim.x + threadIdx.x;
  if (g >= Ng) return;
  float x0 = in[g * 2 + 0], x1 = in[g * 2 + 1];
  float v0 = b[0] + x0 * W[0] + x1 * W[2];
  float v1 = b[1] + x0 * W[1] + x1 * W[3];
  float m  = fmaxf(v0, v1);
  float lse = m + logf(expf(v0 - m) + expf(v1 - m));
  out[g * 2 + 0] = v0 - lse;
  out[g * 2 + 1] = v1 - lse;
}

// ------------------------------ host orchestration ------------------------

static inline size_t alignup(size_t x) { return (x + 255) & ~(size_t)255; }
static inline int cdiv(long long a, int b) { return (int)((a + b - 1) / b); }

extern "C" void kernel_launch(void* const* d_in, const int* in_sizes, int n_in,
                              void* d_out, int out_size, void* d_ws, size_t ws_size,
                              hipStream_t stream) {
  (void)in_sizes; (void)n_in; (void)out_size; (void)ws_size;

  const float* x     = (const float*)d_in[0];
  const int*   ei    = (const int*)d_in[1];
  const int*   batch = (const int*)d_in[2];
  const float* ew    = (const float*)d_in[3];
  const float* w_rel[3]  = { (const float*)d_in[4],  (const float*)d_in[7],  (const float*)d_in[10] };
  const float* b_rel[3]  = { (const float*)d_in[5],  (const float*)d_in[8],  (const float*)d_in[11] };
  const float* w_root[3] = { (const float*)d_in[6],  (const float*)d_in[9],  (const float*)d_in[12] };
  const float* gamma[3]  = { (const float*)d_in[13], (const float*)d_in[15], (const float*)d_in[17] };
  const float* beta[3]   = { (const float*)d_in[14], (const float*)d_in[16], (const float*)d_in[18] };
  const float* w_lin1 = (const float*)d_in[19];
  const float* b_lin1 = (const float*)d_in[20];
  const float* w_lin2 = (const float*)d_in[21];
  const float* b_lin2 = (const float*)d_in[22];
  const float* w_lin3 = (const float*)d_in[23];
  const float* b_lin3 = (const float*)d_in[24];
  float* out = (float*)d_out;

  // workspace layout (256B aligned)
  char* ws = (char*)d_ws;
  size_t off = 0;
  auto take = [&](size_t bytes) { char* p = ws + off; off += alignup(bytes); return p; };
  const long long NFMAX = (long long)Nn * 384;
  float*  buf0   = (float*)take(NFMAX * 4);
  float*  buf1   = (float*)take(NFMAX * 4);
  __bf16* hbf    = (__bf16*)take(NFMAX * 2);
  __bf16* aggbf  = (__bf16*)take(NFMAX * 2);
  __bf16* wrelp  = (__bf16*)take((size_t)256 * 384 * 2);
  __bf16* wrootp = (__bf16*)take((size_t)256 * 384 * 2);
  float*  stats  = (float*)take((size_t)2 * 384 * 4);
  float*  psum   = (float*)take((size_t)Ng * 384 * 4);
  int*    pmax   = (int*)take((size_t)Ng * 384 * 4);
  float*  pcnt   = (float*)take((size_t)Ng * 4);
  float*  z      = (float*)take((size_t)Ng * ZW * 4);
  float*  z1     = (float*)take((size_t)Ng * 128 * 4);
  float*  z2     = (float*)take((size_t)Ng * 2 * 4);

  const int Fin[3]  = { 128, 128, 256 };
  const int Fout[3] = { 128, 256, 384 };
  const int zoff[3] = { 0, 3 * Hdim, 9 * Hdim };
  const int BLK = 256;
  const int mtiles = Nn / 16;  // 3125

  const float* hin = x;     // layer input (f32)
  float* aggbuf = buf0;
  float* outbuf = buf1;

  for (int L = 0; L < 3; ++L) {
    const int K = Fin[L], F = Fout[L];
    const long long nk = (long long)Nn * K;
    const long long nf = (long long)Nn * F;

    // stage bf16 copies of the root input and the aggregation result
    f32_to_bf16_kernel<<<cdiv(nk, BLK), BLK, 0, stream>>>(hin, hbf, nk);
    zero_f32_kernel<<<cdiv(nk, BLK), BLK, 0, stream>>>(aggbuf, nk);
    edge_agg_kernel<<<cdiv((long long)Ne * (K >> 2), BLK), BLK, 0, stream>>>(
        hin, ei, ew, aggbuf, K);
    f32_to_bf16_kernel<<<cdiv(nk, BLK), BLK, 0, stream>>>(aggbuf, aggbf, nk);

    // pack weights into per-lane WMMA-B layout (tiny)
    int packn = K * F;
    pack_w_kernel<<<cdiv(packn, BLK), BLK, 0, stream>>>(w_rel[L], wrelp, K, F);
    pack_w_kernel<<<cdiv(packn, BLK), BLK, 0, stream>>>(w_root[L], wrootp, K, F);

    // fused dual-GEMM (rel + root) + bias + leaky_relu via LDS-blocked WMMA
    dim3 ggrid(cdiv(mtiles, WAVES), F / (16 * NT_PER_BLK));
    gconv_gemm_wmma<<<ggrid, 32 * WAVES, 0, stream>>>(aggbf, hbf, wrelp, wrootp,
                                                      b_rel[L], outbuf, K, F, mtiles);

    // BatchNorm (training-mode, biased var) in place
    bn_stats_kernel<<<F, 256, 0, stream>>>(outbuf, stats, F);
    bn_apply_kernel<<<cdiv(nf, BLK), BLK, 0, stream>>>(outbuf, stats,
                                                       gamma[L], beta[L], F);

    // per-graph max / mean / sum pooling into z
    pool_init_kernel<<<cdiv((long long)Ng * F, BLK), BLK, 0, stream>>>(psum, pmax, pcnt, F);
    pool_acc_kernel<<<cdiv(nf, BLK), BLK, 0, stream>>>(outbuf, batch, psum, pmax, pcnt, F);
    pool_write_kernel<<<cdiv((long long)Ng * F, BLK), BLK, 0, stream>>>(
        psum, pmax, pcnt, z, zoff[L], F);

    // ping-pong: layer output becomes next input; GEMM reads only the bf16
    // staging copies, so the next layer's output may overwrite its own agg.
    hin = outbuf;
    aggbuf = (hin == buf0) ? buf1 : buf0;
    outbuf = aggbuf;
  }

  // FC head
  fc_kernel<<<cdiv((long long)Ng * 128, BLK), BLK, 0, stream>>>(z, w_lin1, b_lin1, z1, ZW, 128, 1);
  fc_kernel<<<cdiv((long long)Ng * 2, BLK), BLK, 0, stream>>>(z1, w_lin2, b_lin2, z2, 128, 2, 1);
  head_final_kernel<<<cdiv(Ng, BLK), BLK, 0, stream>>>(z2, w_lin3, b_lin3, out);
}
